// OCBAB_30356828848472
// MI455X (gfx1250) — compile-verified, matrix operand order checked by reading, not run on hardware
//
#include <hip/hip_runtime.h>
#include <hip/hip_bf16.h>
#include <stdint.h>
#include <stddef.h>

// ---------------- CDNA5 types ----------------
typedef __bf16 bf16t;
typedef __attribute__((ext_vector_type(16))) __bf16 v16bf;
typedef __attribute__((ext_vector_type(8)))  float  v8f;
typedef __attribute__((ext_vector_type(4)))  unsigned u32x4;

// Problem constants (from reference)
#define BATCH 2
#define HH 256
#define WW 256
#define CC 180
#define WS 8
#define OWS 12
#define NH 6
#define HD 30                 // q/k head dim
#define VHD 120               // v head dim (4*C/NH)
#define MD (BATCH*128*128)    // rows after DWT = 32768
#define RWS (BATCH*HH*WW)     // full-res rows = 131072
#define QSCALE 0.18257418583505536f  // 30^-0.5

// Padded K-strides (multiples of 32 elements -> 64B rows, 16B-aligned chunks)
#define LDA_XD  736   // K=720 padded
#define LDA_ACT 192   // K=180 padded
#define LDA_H1  384   // K=360 padded

// ---------------- CDNA5 async-copy path (probe-guarded) ----------------
#if __has_builtin(__builtin_amdgcn_global_load_async_to_lds_b128) && \
    __has_builtin(__builtin_amdgcn_s_wait_asynccnt)
#define USE_ASYNC 1
#else
#define USE_ASYNC 0
#endif

#if USE_ASYNC
typedef int v4i_g __attribute__((vector_size(16)));   // matches builtin pointee
typedef __attribute__((address_space(1))) v4i_g as1_v4i;
typedef __attribute__((address_space(3))) v4i_g as3_v4i;
#endif

__device__ __forceinline__ void async_cp16(const bf16t* g, bf16t* l) {
#if USE_ASYNC
    // GLOBAL_LOAD_ASYNC_TO_LDS_B128: 16B per op, tracked by ASYNCcnt.
    // AS1 pointer = flat addr; AS3 pointer = low 32 bits of flat LDS addr.
    __builtin_amdgcn_global_load_async_to_lds_b128(
        (as1_v4i*)(uintptr_t)g,
        (as3_v4i*)(unsigned)(uintptr_t)l,
        0, 0);
#else
    *(u32x4*)l = *(const u32x4*)g;
#endif
}
__device__ __forceinline__ void async_wait0() {
#if USE_ASYNC
    __builtin_amdgcn_s_wait_asynccnt(0);
#endif
}

// ---------------- helpers ----------------
__device__ __forceinline__ bf16t f2bf(float f) {
    unsigned u = __builtin_bit_cast(unsigned, f);
    unsigned r = u + 0x7FFFu + ((u >> 16) & 1u);   // round-to-nearest-even
    unsigned short h = (unsigned short)(r >> 16);
    return __builtin_bit_cast(bf16t, h);
}
__device__ __forceinline__ float bf2f(bf16t b) {
    unsigned short h = __builtin_bit_cast(unsigned short, b);
    unsigned u = ((unsigned)h) << 16;
    return __builtin_bit_cast(float, u);
}

// 16x16(x32) WMMA operand fragment from LDS (two ds_load_b128 per frag).
// Lanes 0-15 = rows, K-half selected by lane>>4 (16 bf16 per lane).
__device__ __forceinline__ v16bf ld_frag(const bf16t* base, int stride,
                                         int kbase, int lane) {
    const bf16t* p = base + (size_t)(lane & 15) * stride + kbase + ((lane >> 4) << 4);
    union { u32x4 q[2]; v16bf v; } f;
    const u32x4* p4 = (const u32x4*)p;
    f.q[0] = p4[0];
    f.q[1] = p4[1];
    return f.v;
}

__device__ __forceinline__ v8f wmma_bf16(v16bf a, v16bf b, v8f c) {
    return __builtin_amdgcn_wmma_f32_16x16x32_bf16(
        false, a, false, b, (short)0, c, false, false);
}

// ---------------- kernel 0: weight convert f32[K][N] -> bf16 Wt[Npad][ldb] ---
__global__ __launch_bounds__(256)
void k_convw(const float* __restrict__ W, bf16t* __restrict__ Wt,
             int K, int N, int ldb) {
    int n = blockIdx.x;   // grid.x = Npad
    for (int k = threadIdx.x; k < ldb; k += 256) {
        float v = (n < N && k < K) ? W[(size_t)k * N + n] : 0.f;
        Wt[(size_t)n * ldb + k] = f2bf(v);
    }
}

// ---------------- kernel 1: LayerNorm + Haar DWT (fp32 -> bf16, C -> 4C) ----
__global__ __launch_bounds__(128)
void k_ln_dwt(const float* __restrict__ x, const float* __restrict__ w,
              const float* __restrict__ bvec, bf16t* __restrict__ xd) {
    __shared__ float ln4[4][192];
    int bid = blockIdx.x;                 // (b*128+oy)*128+ox
    int b = bid >> 14, rem = bid & 16383;
    int oy = rem >> 7, ox = rem & 127;
    int lane = threadIdx.x & 31, wave = threadIdx.x >> 5;  // 4 waves = 2x2 quad
    int iy = 2 * oy + (wave >> 1), ix = 2 * ox + (wave & 1);
    const float* px = x + (size_t)((b * 256 + iy) * 256 + ix) * CC;

    float sum = 0.f, sq = 0.f, vals[6];
#pragma unroll
    for (int j = 0; j < 6; ++j) {
        int c = lane + j * 32;
        float v = (c < CC) ? px[c] : 0.f;
        vals[j] = v; sum += v; sq += v * v;
    }
#pragma unroll
    for (int s = 16; s > 0; s >>= 1) {
        sum += __shfl_xor(sum, s, 32);
        sq  += __shfl_xor(sq,  s, 32);
    }
    float mu = sum * (1.f / CC);
    float rs = rsqrtf(sq * (1.f / CC) - mu * mu + 1e-5f);
#pragma unroll
    for (int j = 0; j < 6; ++j) {
        int c = lane + j * 32;
        if (c < CC) ln4[wave][c] = (vals[j] - mu) * rs * w[c] + bvec[c];
    }
    __syncthreads();
    bf16t* out = xd + (size_t)bid * LDA_XD;
    for (int c = threadIdx.x; c < CC; c += 128) {
        float a   = ln4[0][c], bb = ln4[1][c];
        float cc2 = ln4[2][c], dd = ln4[3][c];
        out[c]        = f2bf(( a + bb + cc2 + dd) * 0.5f);   // LL
        out[CC + c]   = f2bf((-a - bb + cc2 + dd) * 0.5f);   // LH
        out[2*CC + c] = f2bf((-a + bb - cc2 + dd) * 0.5f);   // HL
        out[3*CC + c] = f2bf(( a - bb - cc2 + dd) * 0.5f);   // HH
    }
    if (threadIdx.x < LDA_XD - 720) out[720 + threadIdx.x] = f2bf(0.f);  // K pad
}

// ---------------- kernel 2: bf16 WMMA GEMM, 128x64 tile, async double-buffer -
// C[M,N] = A[M,K](bf16, row stride lda) @ Bt[N,K](bf16, row stride ldb)^T
// EPI: 0 = bias -> bf16 ; 1 = bias+GELU -> bf16 ; 2 = bias+residual -> f32
template <int EPI>
__global__ __launch_bounds__(256)
void k_gemm(const bf16t* __restrict__ A, int lda,
            const bf16t* __restrict__ Bt, int ldb,
            const float* __restrict__ bias, const float* __restrict__ resid,
            bf16t* __restrict__ Cb, float* __restrict__ Cf,
            int M, int N, int K, int ldc) {
    __shared__ bf16t AsBuf[2][128 * 40];   // 128 rows x 32 K (+pad)
    __shared__ bf16t BsBuf[2][64 * 40];    // 64 cols (N-major) x 32 K (+pad)
    const int tid  = threadIdx.x;
    const int lane = tid & 31, wave = tid >> 5;
    const int wm = wave & 3, wn = wave >> 2;           // 4x2 wave grid
    const int m0 = blockIdx.y * 128, n0 = blockIdx.x * 64;

    // Per-step tile fill: 512 A-chunks + 256 B-chunks of 16B, all in-range
    // (K-strides padded to multiples of 32 elems, B rows padded to Npad).
    const int ar0 = (tid) >> 2,        ac0 = (tid & 3) * 8;
    const int ar1 = (tid + 256) >> 2,  ac1 = ((tid + 256) & 3) * 8;
    auto issue_tile = [&](int k0, int buf) {
        bf16t* As = AsBuf[buf];
        bf16t* Bs = BsBuf[buf];
        async_cp16(A + (size_t)(m0 + ar0) * lda + k0 + ac0, As + ar0 * 40 + ac0);
        async_cp16(A + (size_t)(m0 + ar1) * lda + k0 + ac1, As + ar1 * 40 + ac1);
        async_cp16(Bt + (size_t)(n0 + ar0) * ldb + k0 + ac0, Bs + ar0 * 40 + ac0);
    };

    v8f c00 = {0,0,0,0,0,0,0,0}, c01 = c00, c10 = c00, c11 = c00;
    const int steps = (K + 31) >> 5;
    issue_tile(0, 0);
    for (int s = 0; s < steps; ++s) {
        async_wait0();
        __syncthreads();                       // tile s visible to all waves
        if (s + 1 < steps) issue_tile((s + 1) << 5, (s + 1) & 1);
        const bf16t* As = AsBuf[s & 1];
        const bf16t* Bs = BsBuf[s & 1];
        v16bf a0 = ld_frag(As + (wm * 32) * 40,      40, 0, lane);
        v16bf a1 = ld_frag(As + (wm * 32 + 16) * 40, 40, 0, lane);
        v16bf b0 = ld_frag(Bs + (wn * 32) * 40,      40, 0, lane);
        v16bf b1 = ld_frag(Bs + (wn * 32 + 16) * 40, 40, 0, lane);
        c00 = wmma_bf16(a0, b0, c00);
        c01 = wmma_bf16(a0, b1, c01);
        c10 = wmma_bf16(a1, b0, c10);
        c11 = wmma_bf16(a1, b1, c11);
        __syncthreads();                       // done reading buf (s&1)
    }

#pragma unroll
    for (int tm = 0; tm < 2; ++tm) {
#pragma unroll
        for (int tn = 0; tn < 2; ++tn) {
            const v8f acc = tm ? (tn ? c11 : c10) : (tn ? c01 : c00);
#pragma unroll
            for (int i = 0; i < 8; ++i) {
                int m = m0 + wm * 32 + tm * 16 + i + ((lane >> 4) << 3);
                int n = n0 + wn * 32 + tn * 16 + (lane & 15);
                if (m < M && n < ldc) {
                    float v = 0.f;
                    if (n < N) {
                        v = acc[i] + bias[n];
                        if (EPI == 1) v = 0.5f * v * (1.f + erff(v * 0.70710678118f));
                    }
                    if (EPI == 2) {
                        if (n < N) Cf[(size_t)m * ldc + n] = v + resid[(size_t)m * ldc + n];
                    } else {
                        Cb[(size_t)m * ldc + n] = f2bf(v);  // zero-fills K pads
                    }
                }
            }
        }
    }
}

// ---------------- kernel 3: windowed attention, one (window, head) per block -
__global__ __launch_bounds__(128)
void k_attn(const bf16t* __restrict__ q, const bf16t* __restrict__ k,
            const bf16t* __restrict__ v, const int* __restrict__ rpi,
            const float* __restrict__ rpb, bf16t* __restrict__ o) {
    extern __shared__ char smem[];
    bf16t* qs = (bf16t*)smem;                  // 64 x 40   (d padded to 32)
    bf16t* ks = qs + 64 * 40;                  // 144 x 40
    float* sS = (float*)(ks + 144 * 40);       // 64 x 144  scores (fp32)
    bf16t* ps = (bf16t*)(sS + 64 * 144);       // 64 x 160  probs (K padded)
    bf16t* vs = ps + 64 * 160;                 // 128 x 160 V^T (N padded)

    int blk = blockIdx.x;
    int win = blk / NH, h = blk % NH;
    int b = win >> 8, rem = win & 255;
    int wh = rem >> 4, wwi = rem & 15;
    int tid = threadIdx.x, lane = tid & 31, wave = tid >> 5;

    for (int idx = tid; idx < 64 * 32; idx += 128) {        // q tile (scaled)
        int p = idx >> 5, c = idx & 31;
        int gy = wh * WS + (p >> 3), gx = wwi * WS + (p & 7);
        size_t row = (size_t)((b * 128 + gy) * 128 + gx);
        float val = (c < HD) ? bf2f(q[row * CC + h * HD + c]) * QSCALE : 0.f;
        qs[p * 40 + c] = f2bf(val);
    }
    for (int idx = tid; idx < 144 * 32; idx += 128) {       // k tile (unfolded)
        int n = idx >> 5, c = idx & 31;
        int gy = wh * WS - 4 + n / OWS, gx = wwi * WS - 4 + n % OWS;
        float val = 0.f;
        if (c < HD && gy >= 0 && gy < 128 && gx >= 0 && gx < 128) {
            size_t row = (size_t)((b * 128 + gy) * 128 + gx);
            val = bf2f(k[row * CC + h * HD + c]);
        }
        ks[n * 40 + c] = f2bf(val);
    }
    for (int idx = tid; idx < 128 * 160; idx += 128) {      // v^T tile
        int ch = idx / 160, key = idx % 160;
        float val = 0.f;
        if (ch < VHD && key < 144) {
            int gy = wh * WS - 4 + key / OWS, gx = wwi * WS - 4 + key % OWS;
            if (gy >= 0 && gy < 128 && gx >= 0 && gx < 128) {
                size_t row = (size_t)((b * 128 + gy) * 128 + gx);
                val = bf2f(v[row * (4 * CC) + h * VHD + ch]);
            }
        }
        vs[ch * 160 + key] = f2bf(val);
    }
    __syncthreads();

    // S = q k^T + bias  (wave owns 16 query rows; 9 key tiles)
    v16bf aq = ld_frag(qs + wave * 16 * 40, 40, 0, lane);
    for (int nt = 0; nt < 9; ++nt) {
        v16bf bk = ld_frag(ks + nt * 16 * 40, 40, 0, lane);
        v8f acc = {0,0,0,0,0,0,0,0};
        acc = wmma_bf16(aq, bk, acc);
#pragma unroll
        for (int i = 0; i < 8; ++i) {
            int m = wave * 16 + i + ((lane >> 4) << 3);
            int n = nt * 16 + (lane & 15);
            sS[m * 144 + n] = acc[i] + rpb[rpi[m * 144 + n] * NH + h];
        }
    }
    __syncthreads();

    // softmax over 144 keys (wave32 shuffle reductions)
    for (int r = wave * 16; r < wave * 16 + 16; ++r) {
        float mx = -1e30f;
        for (int c = lane; c < 144; c += 32) mx = fmaxf(mx, sS[r * 144 + c]);
#pragma unroll
        for (int s = 16; s > 0; s >>= 1) mx = fmaxf(mx, __shfl_xor(mx, s, 32));
        float sum = 0.f;
        for (int c = lane; c < 144; c += 32) {
            float e = __expf(sS[r * 144 + c] - mx);
            sS[r * 144 + c] = e; sum += e;
        }
#pragma unroll
        for (int s = 16; s > 0; s >>= 1) sum += __shfl_xor(sum, s, 32);
        float inv = 1.f / sum;
        for (int c = lane; c < 160; c += 32)
            ps[r * 160 + c] = (c < 144) ? f2bf(sS[r * 144 + c] * inv) : f2bf(0.f);
    }
    __syncthreads();

    // O = P @ V : K=144 -> 5 zero-padded steps of 32; N=120 -> 8 tiles
    bf16t* orow = o + (size_t)win * 64 * (4 * CC) + h * VHD;
    for (int nt = 0; nt < 8; ++nt) {
        v8f acc = {0,0,0,0,0,0,0,0};
#pragma unroll
        for (int kt = 0; kt < 5; ++kt) {
            v16bf ap = ld_frag(ps + wave * 16 * 160, 160, kt * 32, lane);
            v16bf bv = ld_frag(vs + nt * 16 * 160, 160, kt * 32, lane);
            acc = wmma_bf16(ap, bv, acc);
        }
#pragma unroll
        for (int i = 0; i < 8; ++i) {
            int m = wave * 16 + i + ((lane >> 4) << 3);
            int n = nt * 16 + (lane & 15);
            if (n < VHD) orow[(size_t)m * (4 * CC) + n] = f2bf(acc[i]);
        }
    }
}

// ---------------- kernel 4: window-reverse + inverse DWT ---------------------
__global__ __launch_bounds__(192)
void k_idwt(const bf16t* __restrict__ o, bf16t* __restrict__ xr) {
    int bid = blockIdx.x;                   // (b*256+y)*256+x
    int b = bid >> 16, rem = bid & 65535;
    int y = rem >> 8, xi = rem & 255;
    int c = threadIdx.x;                    // 0..191 (pads write zero)
    float r = 0.f;
    if (c < CC) {
        int h2 = y >> 1, w2 = xi >> 1;
        int win = b * 256 + (h2 >> 3) * 16 + (w2 >> 3);
        int pix = (h2 & 7) * 8 + (w2 & 7);
        const bf16t* base = o + ((size_t)win * 64 + pix) * (4 * CC) + c;
        float ll = bf2f(base[0]),      lh = bf2f(base[CC]);
        float hl = bf2f(base[2 * CC]), hh = bf2f(base[3 * CC]);
        int dy = y & 1, dx = xi & 1;
        if (!dy && !dx)      r = (ll - lh - hl + hh);
        else if (!dy &&  dx) r = (ll - lh + hl - hh);
        else if ( dy && !dx) r = (ll + lh - hl - hh);
        else                 r = (ll + lh + hl + hh);
        r *= 0.5f;
    }
    xr[(size_t)bid * LDA_ACT + c] = f2bf(r);
}

// ---------------- kernel 5: row LayerNorm (fp32 -> bf16, padded stride) ------
__global__ __launch_bounds__(256)
void k_ln_rows(const float* __restrict__ xin, const float* __restrict__ w,
               const float* __restrict__ bvec, bf16t* __restrict__ out) {
    int lane = threadIdx.x & 31, wave = threadIdx.x >> 5;
    size_t row = (size_t)blockIdx.x * 8 + wave;
    const float* px = xin + row * CC;
    float sum = 0.f, sq = 0.f, vals[6];
#pragma unroll
    for (int j = 0; j < 6; ++j) {
        int c = lane + j * 32;
        float v = (c < CC) ? px[c] : 0.f;
        vals[j] = v; sum += v; sq += v * v;
    }
#pragma unroll
    for (int s = 16; s > 0; s >>= 1) {
        sum += __shfl_xor(sum, s, 32);
        sq  += __shfl_xor(sq,  s, 32);
    }
    float mu = sum * (1.f / CC);
    float rs = rsqrtf(sq * (1.f / CC) - mu * mu + 1e-5f);
#pragma unroll
    for (int j = 0; j < 6; ++j) {
        int c = lane + j * 32;                 // covers 0..191 = LDA_ACT
        float v = (c < CC) ? (vals[j] - mu) * rs * w[c] + bvec[c] : 0.f;
        out[row * LDA_ACT + c] = f2bf(v);
    }
}

// ---------------- host launch ------------------------------------------------
extern "C" void kernel_launch(void* const* d_in, const int* in_sizes, int n_in,
                              void* d_out, int out_size, void* d_ws, size_t ws_size,
                              hipStream_t stream) {
    (void)in_sizes; (void)n_in; (void)out_size; (void)ws_size;
    const float* x       = (const float*)d_in[0];
    const int*   rpi     = (const int*)  d_in[2];
    const float* norm1_w = (const float*)d_in[5];
    const float* norm1_b = (const float*)d_in[6];
    const float* q_w     = (const float*)d_in[9];
    const float* q_b     = (const float*)d_in[10];
    const float* k_w     = (const float*)d_in[11];
    const float* k_b     = (const float*)d_in[12];
    const float* v_w     = (const float*)d_in[13];
    const float* v_b     = (const float*)d_in[14];
    const float* rpb     = (const float*)d_in[15];
    const float* proj_w  = (const float*)d_in[16];
    const float* proj_b  = (const float*)d_in[17];
    const float* norm2_w = (const float*)d_in[18];
    const float* norm2_b = (const float*)d_in[19];
    const float* fc1_w   = (const float*)d_in[20];
    const float* fc1_b   = (const float*)d_in[21];
    const float* fc2_w   = (const float*)d_in[22];
    const float* fc2_b   = (const float*)d_in[23];

    char* ws = (char*)d_ws;
    size_t off = 0;
    auto alloc = [&](size_t bytes) {
        size_t o = off; off += (bytes + 255) & ~(size_t)255; return o;
    };
    size_t regbytes = 2 * ((size_t)RWS * LDA_ACT > (size_t)MD * LDA_XD
                          ? (size_t)RWS * LDA_ACT : (size_t)MD * LDA_XD);
    bf16t* xd   = (bf16t*)(ws + alloc(regbytes));             // reused: o, x1n
    bf16t* qb   = (bf16t*)(ws + alloc((size_t)MD * 180 * 2));
    bf16t* kb   = (bf16t*)(ws + alloc((size_t)MD * 180 * 2));
    bf16t* vb   = (bf16t*)(ws + alloc(regbytes));             // reused: xr
    float* x1   = (float*)(ws + alloc((size_t)RWS * 180 * 4));
    bf16t* h1   = (bf16t*)(ws + alloc((size_t)RWS * LDA_H1 * 2));
    bf16t* qwT  = (bf16t*)(ws + alloc((size_t)192 * LDA_XD * 2));
    bf16t* kwT  = (bf16t*)(ws + alloc((size_t)192 * LDA_XD * 2));
    bf16t* vwT  = (bf16t*)(ws + alloc((size_t)768 * LDA_XD * 2));
    bf16t* pjT  = (bf16t*)(ws + alloc((size_t)192 * LDA_ACT * 2));
    bf16t* f1T  = (bf16t*)(ws + alloc((size_t)384 * LDA_ACT * 2));
    bf16t* f2T  = (bf16t*)(ws + alloc((size_t)192 * LDA_H1 * 2));
    bf16t* ob  = xd;   // attention output (xd dead after v GEMM)
    bf16t* xr  = vb;   // iDWT output (v dead after attention)
    bf16t* x1n = xd;   // LN(x1) (o dead after iDWT)

    // 0) one-time weight convert: f32[K][N] -> bf16 [Npad][ldbK] (transposed)
    k_convw<<<dim3(192), dim3(256), 0, stream>>>(q_w,   qwT, 720, 180, LDA_XD);
    k_convw<<<dim3(192), dim3(256), 0, stream>>>(k_w,   kwT, 720, 180, LDA_XD);
    k_convw<<<dim3(768), dim3(256), 0, stream>>>(v_w,   vwT, 720, 720, LDA_XD);
    k_convw<<<dim3(192), dim3(256), 0, stream>>>(proj_w, pjT, 180, 180, LDA_ACT);
    k_convw<<<dim3(384), dim3(256), 0, stream>>>(fc1_w, f1T, 180, 360, LDA_ACT);
    k_convw<<<dim3(192), dim3(256), 0, stream>>>(fc2_w, f2T, 360, 180, LDA_H1);

    // 1) LN + DWT
    k_ln_dwt<<<dim3(MD), dim3(128), 0, stream>>>(x, norm1_w, norm1_b, xd);
    // 2-4) q, k, v projections (K=720)
    k_gemm<0><<<dim3(3, 256),  dim3(256), 0, stream>>>(xd, LDA_XD, qwT, LDA_XD,
        q_b, nullptr, qb, nullptr, MD, 180, 720, 180);
    k_gemm<0><<<dim3(3, 256),  dim3(256), 0, stream>>>(xd, LDA_XD, kwT, LDA_XD,
        k_b, nullptr, kb, nullptr, MD, 180, 720, 180);
    k_gemm<0><<<dim3(12, 256), dim3(256), 0, stream>>>(xd, LDA_XD, vwT, LDA_XD,
        v_b, nullptr, vb, nullptr, MD, 720, 720, 720);
    // 5) attention: 512 windows x 6 heads
    size_t smem = (size_t)64 * 40 * 2 + 144 * 40 * 2 + 64 * 144 * 4
                + (size_t)64 * 160 * 2 + 128 * 160 * 2;   // ~112 KB dynamic LDS
    k_attn<<<dim3(512 * NH), dim3(128), smem, stream>>>(qb, kb, vb, rpi, rpb, ob);
    // 6) window-reverse + iDWT
    k_idwt<<<dim3(RWS), dim3(192), 0, stream>>>(ob, xr);
    // 7) proj + residual(x) -> x1 (fp32)
    k_gemm<2><<<dim3(3, 1024), dim3(256), 0, stream>>>(xr, LDA_ACT, pjT, LDA_ACT,
        proj_b, x, nullptr, x1, RWS, 180, 180, 180);
    // 8) LN(x1)
    k_ln_rows<<<dim3(RWS / 8), dim3(256), 0, stream>>>(x1, norm2_w, norm2_b, x1n);
    // 9) fc1 + GELU (output padded to 384 cols, zero-filled)
    k_gemm<1><<<dim3(6, 1024), dim3(256), 0, stream>>>(x1n, LDA_ACT, f1T, LDA_ACT,
        fc1_b, nullptr, h1, nullptr, RWS, 360, 180, LDA_H1);
    // 10) fc2 + residual(x1) -> out (fp32)
    k_gemm<2><<<dim3(3, 1024), dim3(256), 0, stream>>>(h1, LDA_H1, f2T, LDA_H1,
        fc2_b, x1, nullptr, (float*)d_out, RWS, 180, 360, 180);
}
